// GCN_Sparse_5308579578416
// MI455X (gfx1250) — matrix operand order repaired and checked by reading, not compiled, not run on hardware
//
#include <hip/hip_runtime.h>

// ---------- problem constants ----------
#define NN     200000      // nodes (divisible by 32)
#define NA     300         // anchors
#define NAP    320         // anchors padded to multiple of 32
#define NF     128         // input features
#define NH     128         // hidden
#define NC     32          // classes
#define VSMALL 1e-12f
#define CHUNK  4000        // split-K node chunk
#define NCHUNK (NN / CHUNK)  // 50
#define CSBLK  ((NN + 255) / 256)  // colsum partial blocks (782)

typedef __attribute__((ext_vector_type(16))) __bf16          v16bf;
typedef __attribute__((ext_vector_type(8)))  float           v8f;
typedef __attribute__((ext_vector_type(8)))  unsigned short  u16x8;

struct FragPair { u16x8 lo, hi; };

__device__ __forceinline__ v16bf pack_frag(u16x8 lo, u16x8 hi) {
  FragPair f{lo, hi};
  return __builtin_bit_cast(v16bf, f);   // 32B == 32B; loads land in frag regs
}

__device__ __forceinline__ unsigned short f2bf(float f) {
  unsigned u = __float_as_uint(f);
  u += 0x7fffu + ((u >> 16) & 1u);       // round-to-nearest-even
  return (unsigned short)(u >> 16);
}

// tiled layout for support: 16x16 tiles, column-major inside tile, tiles row-major
__device__ __forceinline__ size_t tiled_idx(int row, int col, int Nn) {
  return ((size_t)(row >> 4) * (Nn >> 4) + (col >> 4)) * 256 + (col & 15) * 16 + (row & 15);
}

// ---------- fragment loaders (contiguous k order k = 16*(lane/16) + s, wave32) ----------
// A (16x32) from row-major [M x K]:
//   lanes 0-15 : M=lane,    K=0..15
//   lanes 16-31: M=lane-16, K=16..31
__device__ __forceinline__ v16bf load_a_frag(const unsigned short* __restrict__ A,
                                             int row0, size_t ldk, int k0, int lane) {
  int m  = lane & 15;
  int kb = (lane & 16) ? 16 : 0;
  const unsigned short* p = A + (size_t)(row0 + m) * ldk + k0 + kb;
  return pack_frag(*(const u16x8*)(p), *(const u16x8*)(p + 8));
}

// B (32x16) from B^T stored row-major [N x K]:
//   lanes 0-15 : N=lane, K=0..15 ; lanes 16-31: N=lane-16, K=16..31
__device__ __forceinline__ v16bf load_bt_frag(const unsigned short* __restrict__ Bt,
                                              int col0, size_t ldk, int k0, int lane) {
  int n  = lane & 15;
  int kb = (lane & 16) ? 16 : 0;
  const unsigned short* p = Bt + (size_t)(col0 + n) * ldk + k0 + kb;
  return pack_frag(*(const u16x8*)(p), *(const u16x8*)(p + 8));
}

// B (32x16) from tiled support layout (K dim = nodes)
__device__ __forceinline__ v16bf load_b_tiled(const unsigned short* __restrict__ S,
                                              int n0, int d, int dBase, int lane) {
  int n    = lane & 15;
  int node = n0 + ((lane & 16) ? 16 : 0);          // multiple of 16
  const unsigned short* p =
      S + ((size_t)(node >> 4) * (d >> 4) + (dBase >> 4)) * 256 + n * 16;
  return pack_frag(*(const u16x8*)(p), *(const u16x8*)(p + 8));
}

// ---------- prep kernels ----------
__global__ void cvt_bf16_kernel(const float* __restrict__ in,
                                unsigned short* __restrict__ out, int n) {
  int i = blockIdx.x * blockDim.x + threadIdx.x;
  if (i < n) out[i] = f2bf(in[i]);
}

// W [K x Nn] f32 -> Wt [Nn x K] bf16
__global__ void wtrans_kernel(const float* __restrict__ W,
                              unsigned short* __restrict__ Wt, int K, int Nn) {
  int i = blockIdx.x * blockDim.x + threadIdx.x;
  if (i >= K * Nn) return;
  int k = i % K, n = i / K;
  Wt[i] = f2bf(W[(size_t)k * Nn + n]);
}

// per-node: rowsum + bf16 copy of adj padded NA->NAP (row-major, for gemm3 A)
__global__ void adj_prep_kernel(const float* __restrict__ adj,
                                unsigned short* __restrict__ adjB,
                                float* __restrict__ rowsum) {
  int n = blockIdx.x * blockDim.x + threadIdx.x;
  if (n >= NN) return;
  const float4* row = (const float4*)(adj + (size_t)n * NA);
  unsigned short* o = adjB + (size_t)n * NAP;
  float s = 0.f;
#pragma unroll 1
  for (int j = 0; j < NA / 4; ++j) {
    float4 v = row[j];
    s += v.x + v.y + v.z + v.w;
    o[4 * j + 0] = f2bf(v.x);
    o[4 * j + 1] = f2bf(v.y);
    o[4 * j + 2] = f2bf(v.z);
    o[4 * j + 3] = f2bf(v.w);
  }
#pragma unroll
  for (int j = NA; j < NAP; ++j) o[j] = 0;
  rowsum[n] = s;
}

// LDS-tiled transpose: adj f32 [NN x NA] -> adjT bf16 [NAP x NN] (zero-pad anchors)
__global__ void adjT_kernel(const float* __restrict__ adj,
                            unsigned short* __restrict__ adjT) {
  __shared__ unsigned short t[32][33];
  int n0 = blockIdx.x * 32;
  int a0 = blockIdx.y * 32;
  int tx = threadIdx.x & 31, ty = threadIdx.x >> 5;   // 256 threads: ty 0..7
#pragma unroll
  for (int i = ty; i < 32; i += 8) {
    int a = a0 + tx;
    float v = (a < NA) ? adj[(size_t)(n0 + i) * NA + a] : 0.f;
    t[tx][i] = f2bf(v);                               // t[a_local][n_local]
  }
  __syncthreads();
#pragma unroll
  for (int i = ty; i < 32; i += 8)
    adjT[(size_t)(a0 + i) * NN + n0 + tx] = t[i][tx];
}

// per-block column partial sums (deterministic; coalesced rows)
__global__ void colsum_kernel(const float* __restrict__ adj,
                              float* __restrict__ colpart) {
  int a = threadIdx.x;               // blockDim.x == NAP (320)
  int b = blockIdx.x;
  float s = 0.f;
  if (a < NA) {
    int base = b * 256;
    int end  = base + 256 < NN ? base + 256 : NN;
    for (int i = base; i < end; ++i) s += adj[(size_t)i * NA + a];
  }
  colpart[(size_t)b * NAP + a] = s;
}

// ordered reduce of column partials
__global__ void colsum_reduce_kernel(const float* __restrict__ colpart,
                                     float* __restrict__ colsum) {
  int a = blockIdx.x * blockDim.x + threadIdx.x;
  if (a >= NAP) return;
  float s = 0.f;
  for (int b = 0; b < CSBLK; ++b) s += colpart[(size_t)b * NAP + a];
  colsum[a] = s;
}

// ---------- main WMMA GEMM: C[M x Nn] = A[M x K] * Bt[Nn x K]^T ----------
// 2 row-tiles per wave (M must be a multiple of 32); B fragment shared.
// EPI 0: store bf16 in tiled layout (support)
// EPI 1: v /= max(rowsum[row],eps); relu; store bf16 row-major
// EPI 2: v /= max(rowsum[row],eps); store f32 row-major
template <int EPI>
__device__ __forceinline__ void gemm_epilogue(const v8f& acc, int row0, int col,
                                              void* out,
                                              const float* __restrict__ rowsum,
                                              int Nn, int lane) {
  int rbase = row0 + ((lane & 16) ? 8 : 0);
#pragma unroll
  for (int r = 0; r < 8; ++r) {
    int row = rbase + r;
    float v = acc[r];
    if (EPI == 0) {
      ((unsigned short*)out)[tiled_idx(row, col, Nn)] = f2bf(v);
    } else {
      v = v / fmaxf(rowsum[row], VSMALL);
      if (EPI == 1) {
        v = fmaxf(v, 0.f);
        ((unsigned short*)out)[(size_t)row * Nn + col] = f2bf(v);
      } else {
        ((float*)out)[(size_t)row * Nn + col] = v;
      }
    }
  }
}

template <int EPI>
__global__ void gemm_bf16_kernel(const unsigned short* __restrict__ A,
                                 const unsigned short* __restrict__ Bt,
                                 void* __restrict__ out,
                                 const float* __restrict__ rowsum,
                                 int M, int Nn, int K) {
  int lane = threadIdx.x & 31;
  int wave = threadIdx.x >> 5;
  int ntiles = Nn >> 4;
  int tiles  = (M >> 5) * ntiles;      // 32-row blocks
  int t = blockIdx.x * 8 + wave;
  if (t >= tiles) return;
  int tm = t / ntiles, tn = t % ntiles;
  int row0 = tm << 5, col0 = tn << 4;

  v8f acc0 = {0.f, 0.f, 0.f, 0.f, 0.f, 0.f, 0.f, 0.f};
  v8f acc1 = acc0;
  for (int k0 = 0; k0 < K; k0 += 32) {
    v16bf b  = load_bt_frag(Bt, col0, K, k0, lane);
    v16bf a0 = load_a_frag(A, row0,      K, k0, lane);
    v16bf a1 = load_a_frag(A, row0 + 16, K, k0, lane);
    acc0 = __builtin_amdgcn_wmma_f32_16x16x32_bf16(false, a0, false, b,
                                                   (short)0, acc0, false, false);
    acc1 = __builtin_amdgcn_wmma_f32_16x16x32_bf16(false, a1, false, b,
                                                   (short)0, acc1, false, false);
  }
  int col = col0 + (lane & 15);
  gemm_epilogue<EPI>(acc0, row0,      col, out, rowsum, Nn, lane);
  gemm_epilogue<EPI>(acc1, row0 + 16, col, out, rowsum, Nn, lane);
}

// ---------- split-K WMMA: mid_part[chunk] = adjT(chunk) @ support(chunk) ----------
// A = adjT [NAP x NN] row-major (contiguous in node); B = tiled support.
// Deterministic: each chunk writes its own partial plane; no atomics.
__global__ void gemm_mid_kernel(const unsigned short* __restrict__ adjT,
                                const unsigned short* __restrict__ S,
                                float* __restrict__ mid_part, int d) {
  int lane = threadIdx.x & 31;
  int wave = threadIdx.x >> 5;
  int ntiles = d >> 4;
  int pairs  = (NAP >> 5) * ntiles;    // 10 * ntiles (32-anchor blocks)
  int t = blockIdx.x * 8 + wave;
  if (t >= pairs) return;
  int tm = t / ntiles, tn = t % ntiles;
  int aBase = tm << 5, dBase = tn << 4;
  int n0   = blockIdx.y * CHUNK;
  int nEnd = n0 + CHUNK < NN ? n0 + CHUNK : NN;

  v8f acc0 = {0.f, 0.f, 0.f, 0.f, 0.f, 0.f, 0.f, 0.f};
  v8f acc1 = acc0;
  for (int n = n0; n < nEnd; n += 32) {
    v16bf b  = load_b_tiled(S, n, d, dBase, lane);
    v16bf a0 = load_a_frag(adjT, aBase,      (size_t)NN, n, lane);
    v16bf a1 = load_a_frag(adjT, aBase + 16, (size_t)NN, n, lane);
    acc0 = __builtin_amdgcn_wmma_f32_16x16x32_bf16(false, a0, false, b,
                                                   (short)0, acc0, false, false);
    acc1 = __builtin_amdgcn_wmma_f32_16x16x32_bf16(false, a1, false, b,
                                                   (short)0, acc1, false, false);
  }
  float* plane = mid_part + (size_t)blockIdx.y * NAP * d;
  int col = dBase + (lane & 15);
  int rb0 = aBase + ((lane & 16) ? 8 : 0);
#pragma unroll
  for (int r = 0; r < 8; ++r) {
    plane[(size_t)(rb0 + r) * d + col]      = acc0[r];
    plane[(size_t)(rb0 + 16 + r) * d + col] = acc1[r];
  }
}

// midT[d x NAP] bf16 = (sum_chunks mid_part / colsum)^T, ordered reduction
__global__ void mid_finalize_kernel(const float* __restrict__ mid_part,
                                    const float* __restrict__ colsum,
                                    unsigned short* __restrict__ midT, int d) {
  int i = blockIdx.x * blockDim.x + threadIdx.x;
  if (i >= NAP * d) return;
  int a = i % NAP, dc = i / NAP;
  float s = 0.f;
  for (int c = 0; c < NCHUNK; ++c)
    s += mid_part[(size_t)c * NAP * d + (size_t)a * d + dc];
  float v = (a < NA) ? s / fmaxf(colsum[a], VSMALL) : 0.f;
  midT[i] = f2bf(v);
}

// ---------- host-side orchestration ----------
extern "C" void kernel_launch(void* const* d_in, const int* in_sizes, int n_in,
                              void* d_out, int out_size, void* d_ws, size_t ws_size,
                              hipStream_t stream) {
  const float* x   = (const float*)d_in[0];  // [NN x NF]
  const float* adj = (const float*)d_in[1];  // [NN x NA]
  const float* W1  = (const float*)d_in[2];  // [NF x NH]
  const float* W2  = (const float*)d_in[3];  // [NH x NH]
  const float* W3  = (const float*)d_in[4];  // [NH x NC]

  // workspace carve-up (256B aligned)
  char* base = (char*)d_ws;
  size_t off = 0;
  auto carve = [&](size_t bytes) {
    char* p = base + off;
    off = (off + bytes + 255) & ~(size_t)255;
    return p;
  };
  unsigned short* adjB    = (unsigned short*)carve((size_t)NN * NAP * 2);  // row-major
  unsigned short* adjT    = (unsigned short*)carve((size_t)NAP * NN * 2);  // anchor-major
  unsigned short* bufA    = (unsigned short*)carve((size_t)NN * NF * 2);
  unsigned short* bufB    = (unsigned short*)carve((size_t)NN * NF * 2);
  unsigned short* supp    = (unsigned short*)carve((size_t)NN * NF * 2);   // tiled layout
  float*          rowsum  = (float*)carve((size_t)NN * 4);
  float*          colsum  = (float*)carve((size_t)NAP * 4);
  float*          colpart = (float*)carve((size_t)CSBLK * NAP * 4);
  float*          midpart = (float*)carve((size_t)NCHUNK * NAP * NH * 4);
  unsigned short* midT    = (unsigned short*)carve((size_t)NH * NAP * 2);
  unsigned short* Wt1     = (unsigned short*)carve((size_t)NF * NH * 2);
  unsigned short* Wt2     = (unsigned short*)carve((size_t)NH * NH * 2);
  unsigned short* Wt3     = (unsigned short*)carve((size_t)NH * NC * 2);

  const int TPB = 256;

  // prep: bf16 conversions, transposed adj, deterministic normalizers
  cvt_bf16_kernel<<<(NN * NF + TPB - 1) / TPB, TPB, 0, stream>>>(x, bufA, NN * NF);
  wtrans_kernel<<<(NF * NH + TPB - 1) / TPB, TPB, 0, stream>>>(W1, Wt1, NF, NH);
  wtrans_kernel<<<(NH * NH + TPB - 1) / TPB, TPB, 0, stream>>>(W2, Wt2, NH, NH);
  wtrans_kernel<<<(NH * NC + TPB - 1) / TPB, TPB, 0, stream>>>(W3, Wt3, NH, NC);
  adj_prep_kernel<<<(NN + TPB - 1) / TPB, TPB, 0, stream>>>(adj, adjB, rowsum);
  adjT_kernel<<<dim3(NN / 32, NAP / 32), TPB, 0, stream>>>(adj, adjT);
  colsum_kernel<<<CSBLK, NAP, 0, stream>>>(adj, colpart);
  colsum_reduce_kernel<<<(NAP + TPB - 1) / TPB, TPB, 0, stream>>>(colpart, colsum);

  const unsigned short* Hin[3]  = {bufA, bufB, bufA};
  const unsigned short* Wt[3]   = {Wt1, Wt2, Wt3};
  const int             Kin[3]  = {NF, NH, NH};
  const int             Dw[3]   = {NH, NH, NC};
  void*                 Hout[3] = {bufB, bufA, d_out};

  for (int l = 0; l < 3; ++l) {
    int d = Dw[l];
    int tiles = (NN >> 5) * (d >> 4);
    // support = H @ W  (tiled bf16 output)
    gemm_bf16_kernel<0><<<(tiles + 7) / 8, TPB, 0, stream>>>(
        Hin[l], Wt[l], supp, nullptr, NN, d, Kin[l]);
    // mid partials = adjT @ support per node chunk (deterministic split-K)
    int pairs = (NAP >> 5) * (d >> 4);
    gemm_mid_kernel<<<dim3((pairs + 7) / 8, NCHUNK), TPB, 0, stream>>>(
        adjT, supp, midpart, d);
    // midT = (ordered-sum(mid partials) / colsum)^T in bf16
    mid_finalize_kernel<<<(NAP * d + TPB - 1) / TPB, TPB, 0, stream>>>(
        midpart, colsum, midT, d);
    // out = (adjB @ midT) / rowsum  (+relu for layers 0,1; f32 final)
    if (l < 2)
      gemm_bf16_kernel<1><<<(tiles + 7) / 8, TPB, 0, stream>>>(
          adjB, midT, Hout[l], rowsum, NN, d, NAP);
    else
      gemm_bf16_kernel<2><<<(tiles + 7) / 8, TPB, 0, stream>>>(
          adjB, midT, Hout[l], rowsum, NN, d, NAP);
  }
}
